// GML_78323023610155
// MI455X (gfx1250) — matrix operand order repaired and checked by reading, not compiled, not run on hardware
//
#include <hip/hip_runtime.h>

#define BB 64      // batch
#define NZ 512     // z rows
#define NY 256     // y rows
#define DD 256     // feature dim
#define NF 768     // NZ + NY
#define NS 512     // active Sinkhorn size
#define GAMMA 10.0f
#define MARGIN 0.1f

typedef __attribute__((ext_vector_type(16))) _Float16 v16h;
typedef __attribute__((ext_vector_type(8)))  _Float16 h8v;
typedef __attribute__((ext_vector_type(8)))  float    f8v;
typedef __attribute__((ext_vector_type(8)))  float    v8f;

static __device__ __forceinline__ float wave_sum32(float v) {
#pragma unroll
  for (int off = 16; off >= 1; off >>= 1) v += __shfl_xor(v, off, 32);
  return v;
}

// A fragment for v_wmma_f32_16x16x32_f16: lane holds row M=l%16,
// K = kbase..kbase+7 (elems 0-7) and kbase+16..kbase+23 (elems 8-15), kbase = (l/16)*8.
static __device__ __forceinline__ v16h load_a_frag(const _Float16* p) {
  h8v c0 = *(const h8v*)(p);
  h8v c1 = *(const h8v*)(p + 16);
  v16h r;
#pragma unroll
  for (int i = 0; i < 8; ++i) { r[i] = c0[i]; r[i + 8] = c1[i]; }
  return r;
}

// ---------------- init ----------------
__global__ void zero_kernel(float* __restrict__ sums, float* __restrict__ out, int nsum) {
  int i = blockIdx.x * blockDim.x + threadIdx.x;
  if (i < nsum) sums[i] = 0.0f;
  if (i < 2) out[i] = 0.0f;
}

// ---------------- row L2-normalize + f16 hi/lo split ----------------
__global__ void normalize_rows(const float* __restrict__ z, const float* __restrict__ y,
                               _Float16* __restrict__ oh, _Float16* __restrict__ ol) {
  int wid  = (blockIdx.x * blockDim.x + threadIdx.x) >> 5;
  int lane = threadIdx.x & 31;
  int b = wid / NF, r = wid % NF;
  const float* src = (r < NZ) ? (z + ((size_t)b * NZ + r) * DD)
                              : (y + ((size_t)b * NY + (r - NZ)) * DD);
  float vals[8];
  float ss = 0.f;
#pragma unroll
  for (int i = 0; i < 8; ++i) {
    float v = src[lane + 32 * i];
    vals[i] = v; ss += v * v;
  }
  ss = wave_sum32(ss);
  float inv = 1.0f / fmaxf(sqrtf(ss), 1e-12f);
  size_t base = ((size_t)b * NF + r) * DD + lane;
#pragma unroll
  for (int i = 0; i < 8; ++i) {
    float v = vals[i] * inv;
    _Float16 h = (_Float16)v;
    oh[base + 32 * i] = h;
    ol[base + 32 * i] = (_Float16)(v - (float)h);
  }
}

// ---------------- sim GEMM: exp(gamma * f1n @ f2n^T), WMMA f16 hi/lo split ----------------
__global__ void sim_gemm(const _Float16* __restrict__ f1h, const _Float16* __restrict__ f1l,
                         const _Float16* __restrict__ f2h, const _Float16* __restrict__ f2l,
                         float* __restrict__ simc, float* __restrict__ rowsum,
                         float* __restrict__ colsum) {
  int tn = blockIdx.x, tm = blockIdx.y, b = blockIdx.z;
  if (tn >= NS / 16 && tm >= NS / 16) return;  // unused corner
  int lane = threadIdx.x;
  int half = lane >> 4, lm = lane & 15;
  const _Float16* arh = f1h + ((size_t)b * NF + tn * 16 + lm) * DD;
  const _Float16* arl = f1l + ((size_t)b * NF + tn * 16 + lm) * DD;
  const _Float16* brh = f2h + ((size_t)b * NF + tm * 16 + lm) * DD;
  const _Float16* brl = f2l + ((size_t)b * NF + tm * 16 + lm) * DD;
  v8f acc = {};
#pragma unroll
  for (int k = 0; k < DD; k += 32) {
    int ka = k + half * 8;
    int kb = k + half * 8;
    v16h Ah = load_a_frag(arh + ka);
    v16h Al = load_a_frag(arl + ka);
    v16h Bh = load_a_frag(brh + kb);
    v16h Bl = load_a_frag(brl + kb);
    acc = __builtin_amdgcn_wmma_f32_16x16x32_f16(false, Ah, false, Bh, (short)0, acc, false, false);
    acc = __builtin_amdgcn_wmma_f32_16x16x32_f16(false, Ah, false, Bl, (short)0, acc, false, false);
    acc = __builtin_amdgcn_wmma_f32_16x16x32_f16(false, Al, false, Bh, (short)0, acc, false, false);
  }
  float rvals[8];
  float cpart = 0.f;
#pragma unroll
  for (int r = 0; r < 8; ++r) {
    float e = __expf(GAMMA * acc[r]);
    rvals[r] = e;
    cpart += e;
  }
  if (tn < NS / 16 && tm < NS / 16) {  // store active block: n = tn*16 + M, m = tm*16 + lm
    float* dst = simc + ((size_t)b * NS + tn * 16 + half * 8) * NS + tm * 16 + lm;
#pragma unroll
    for (int r = 0; r < 8; ++r) dst[(size_t)r * NS] = rvals[r];
  }
  if (tn < NS / 16) {  // rowsum[n] += sum over 16 cols of this tile
#pragma unroll
    for (int r = 0; r < 8; ++r) {
      float v = rvals[r];
#pragma unroll
      for (int off = 1; off < 16; off <<= 1) v += __shfl_xor(v, off, 32);
      if (lm == 0) atomicAdd(&rowsum[b * NS + tn * 16 + half * 8 + r], v);
    }
  }
  if (tm < NS / 16) {  // colsum[m] += sum over 16 rows of this tile
    float c = cpart + __shfl_xor(cpart, 16, 32);
    if (half == 0) atomicAdd(&colsum[b * NS + tm * 16 + lm], c);
  }
}

// ---------------- sim -> cost (in place): C = 1 - s/(rs[n]+cs[m]-s+1e-8) ----------------
__global__ void cost_convert(float* __restrict__ C, const float* __restrict__ rs,
                             const float* __restrict__ cs) {
  size_t i = (size_t)blockIdx.x * blockDim.x + threadIdx.x;
  int m = (int)(i & (NS - 1));
  int n = (int)((i >> 9) & (NS - 1));
  int b = (int)(i >> 18);
  float s = C[i];
  float denom = rs[b * NS + n] + cs[b * NS + m] - s + 1e-8f;
  C[i] = 1.0f - s / denom;
}

// ---------------- softmin over columns m (wave per row n) ----------------
__global__ void softmin_rows(const float* __restrict__ C, const float* __restrict__ g,
                             float* __restrict__ out, float eps, int use_h) {
  int wid  = (blockIdx.x * blockDim.x + threadIdx.x) >> 5;
  int lane = threadIdx.x & 31;
  int b = wid / NS, n = wid % NS;
  const float* row = C + ((size_t)b * NS + n) * NS;
  const float* gb  = use_h ? (g + b * NS) : nullptr;
  float inv = 1.0f / eps;
  float mx = -3.4e38f, s = 0.f;
#pragma unroll 4
  for (int i = 0; i < NS / 32; ++i) {
    int m = lane + 32 * i;
    float v = -row[m] * inv;
    if (use_h) v += gb[m] * inv;
    float mn = fmaxf(mx, v);
    s = s * __expf(mx - mn) + __expf(v - mn);
    mx = mn;
  }
#pragma unroll
  for (int off = 16; off >= 1; off >>= 1) {
    float mo = __shfl_xor(mx, off, 32);
    float so = __shfl_xor(s, off, 32);
    float mn = fmaxf(mx, mo);
    s = s * __expf(mx - mn) + so * __expf(mo - mn);
    mx = mn;
  }
  if (lane == 0) out[b * NS + n] = -eps * (mx + __logf(s));
}

// ---------------- softmin over rows n (thread per column m, h staged in LDS) ----------------
__global__ void softmin_cols(const float* __restrict__ C, const float* __restrict__ f,
                             float* __restrict__ out, float eps, int use_h) {
  __shared__ float sh[NS];
  int b = blockIdx.x / (NS / 256);
  int m = (blockIdx.x % (NS / 256)) * 256 + threadIdx.x;
  float inv = 1.0f / eps;
  for (int i = threadIdx.x; i < NS; i += 256)
    sh[i] = use_h ? f[b * NS + i] * inv : 0.0f;
  __syncthreads();
  const float* col = C + (size_t)b * NS * NS + m;
  float mx = -3.4e38f, s = 0.f;
  for (int n = 0; n < NS; ++n) {
    float v = sh[n] - col[(size_t)n * NS] * inv;
    float mn = fmaxf(mx, v);
    s = s * __expf(mx - mn) + __expf(v - mn);
    mx = mn;
  }
  out[b * NS + m] = -eps * (mx + __logf(s));
}

// ---------------- symmetrized half-step ----------------
__global__ void avg_update(float* __restrict__ fp, const float* __restrict__ ft,
                           float* __restrict__ gp, const float* __restrict__ gt) {
  int i = blockIdx.x * blockDim.x + threadIdx.x;
  fp[i] = 0.5f * (fp[i] + ft[i]);
  gp[i] = 0.5f * (gp[i] + gt[i]);
}

// ---------------- final reduction: mean_b (sum f + sum g) ----------------
__global__ void final_reduce(const float* __restrict__ fv, const float* __restrict__ gv,
                             float* __restrict__ out, float scale) {
  __shared__ float red[8];
  int i = blockIdx.x * 256 + threadIdx.x;
  float v = fv[i] + gv[i];
  v = wave_sum32(v);
  int w = threadIdx.x >> 5;
  if ((threadIdx.x & 31) == 0) red[w] = v;
  __syncthreads();
  if (threadIdx.x == 0) {
    float t = 0.f;
#pragma unroll
    for (int j = 0; j < 8; ++j) t += red[j];
    atomicAdd(out, t * scale);
  }
}

// ---------------- hinge: sum relu(margin - y1 @ y2^T), in-register hi/lo split ----------------
__global__ void hinge_gemm(const float* __restrict__ y1, const float* __restrict__ y2,
                           float* __restrict__ out) {
  int tn = blockIdx.x, tm = blockIdx.y, b = blockIdx.z;
  int lane = threadIdx.x;
  int half = lane >> 4, lm = lane & 15;
  const float* arow = y1 + ((size_t)b * NY + tn * 16 + lm) * DD;
  const float* brow = y2 + ((size_t)b * NY + tm * 16 + lm) * DD;
  v8f acc = {};
#pragma unroll
  for (int k = 0; k < DD; k += 32) {
    int ka = k + half * 8;
    int kb = k + half * 8;
    f8v a0 = *(const f8v*)(arow + ka);
    f8v a1 = *(const f8v*)(arow + ka + 16);
    f8v b0 = *(const f8v*)(brow + kb);
    f8v b1 = *(const f8v*)(brow + kb + 16);
    v16h Ah, Al, Bh, Bl;
#pragma unroll
    for (int i = 0; i < 8; ++i) {
      _Float16 h0 = (_Float16)a0[i];
      _Float16 h1 = (_Float16)a1[i];
      Ah[i] = h0;     Al[i] = (_Float16)(a0[i] - (float)h0);
      Ah[i + 8] = h1; Al[i + 8] = (_Float16)(a1[i] - (float)h1);
      _Float16 p0 = (_Float16)b0[i];
      _Float16 p1 = (_Float16)b1[i];
      Bh[i] = p0;     Bl[i] = (_Float16)(b0[i] - (float)p0);
      Bh[i + 8] = p1; Bl[i + 8] = (_Float16)(b1[i] - (float)p1);
    }
    acc = __builtin_amdgcn_wmma_f32_16x16x32_f16(false, Ah, false, Bh, (short)0, acc, false, false);
    acc = __builtin_amdgcn_wmma_f32_16x16x32_f16(false, Ah, false, Bl, (short)0, acc, false, false);
    acc = __builtin_amdgcn_wmma_f32_16x16x32_f16(false, Al, false, Bh, (short)0, acc, false, false);
  }
  float s = 0.f;
#pragma unroll
  for (int r = 0; r < 8; ++r) s += fmaxf(MARGIN - acc[r], 0.0f);
  s = wave_sum32(s);
  if (lane == 0) atomicAdd(out + 1, s);
}

extern "C" void kernel_launch(void* const* d_in, const int* in_sizes, int n_in,
                              void* d_out, int out_size, void* d_ws, size_t ws_size,
                              hipStream_t stream) {
  (void)in_sizes; (void)n_in; (void)out_size; (void)ws_size;
  const float* z1 = (const float*)d_in[0];
  const float* y1 = (const float*)d_in[1];
  const float* z2 = (const float*)d_in[2];
  const float* y2 = (const float*)d_in[3];
  float* out = (float*)d_out;

  char* ws = (char*)d_ws;
  size_t o = 0;
  const size_t featH = (size_t)BB * NF * DD * sizeof(_Float16);  // 25.2 MB each
  _Float16* f1h = (_Float16*)(ws + o); o += featH;
  _Float16* f1l = (_Float16*)(ws + o); o += featH;
  _Float16* f2h = (_Float16*)(ws + o); o += featH;
  _Float16* f2l = (_Float16*)(ws + o); o += featH;
  float* simC   = (float*)(ws + o);    o += (size_t)BB * NS * NS * sizeof(float);  // 67 MB
  float* rowsum = (float*)(ws + o);    o += (size_t)BB * NS * sizeof(float);
  float* colsum = (float*)(ws + o);    o += (size_t)BB * NS * sizeof(float);
  float* fpot   = (float*)(ws + o);    o += (size_t)BB * NS * sizeof(float);
  float* gpot   = (float*)(ws + o);    o += (size_t)BB * NS * sizeof(float);
  float* ftmp   = (float*)(ws + o);    o += (size_t)BB * NS * sizeof(float);
  float* gtmp   = (float*)(ws + o);    o += (size_t)BB * NS * sizeof(float);

  const int NV = BB * NS;  // 32768

  // 1) zero accumulators (rowsum,colsum contiguous) and outputs
  zero_kernel<<<(2 * NV + 255) / 256, 256, 0, stream>>>(rowsum, out, 2 * NV);

  // 2) normalize + hi/lo split
  normalize_rows<<<BB * NF / 8, 256, 0, stream>>>(z1, y1, f1h, f1l);
  normalize_rows<<<BB * NF / 8, 256, 0, stream>>>(z2, y2, f2h, f2l);

  // 3) sim GEMM (full 768x768 minus corner), store active 512x512 block, row/col sums
  sim_gemm<<<dim3(NF / 16, NF / 16, BB), 32, 0, stream>>>(f1h, f1l, f2h, f2l,
                                                          simC, rowsum, colsum);

  // 4) sim -> cost in place
  cost_convert<<<(int)(((size_t)BB * NS * NS) / 256), 256, 0, stream>>>(simC, rowsum, colsum);

  // 5) Sinkhorn (geomloss epsilon schedule for p=2, diam=3, blur=0.05, scaling=0.5)
  const float eps_list[8] = {9.0f, 9.0f, 2.25f, 0.5625f, 0.140625f,
                             0.03515625f, 0.0087890625f, 0.0025f};
  const int GR = NV / 8;           // wave-per-row blocks of 256
  const int GC = BB * (NS / 256);  // column blocks
  softmin_rows<<<GR, 256, 0, stream>>>(simC, nullptr, fpot, eps_list[0], 0);
  softmin_cols<<<GC, 256, 0, stream>>>(simC, nullptr, gpot, eps_list[0], 0);
  for (int it = 0; it < 8; ++it) {
    float eps = eps_list[it];
    softmin_rows<<<GR, 256, 0, stream>>>(simC, gpot, ftmp, eps, 1);
    softmin_cols<<<GC, 256, 0, stream>>>(simC, fpot, gtmp, eps, 1);
    avg_update<<<NV / 256, 256, 0, stream>>>(fpot, ftmp, gpot, gtmp);
  }
  // final extrapolation at eps = blur^2 (uses pre-update potentials simultaneously)
  softmin_rows<<<GR, 256, 0, stream>>>(simC, gpot, ftmp, eps_list[7], 1);
  softmin_cols<<<GC, 256, 0, stream>>>(simC, fpot, gtmp, eps_list[7], 1);

  // 6) scon_cost = mean_b (sum_n f + sum_m g)
  final_reduce<<<NV / 256, 256, 0, stream>>>(ftmp, gtmp, out, 1.0f / (float)BB);

  // 7) hinge cost
  hinge_gemm<<<dim3(NY / 16, NY / 16, BB), 32, 0, stream>>>(y1, y2, out);
}
